// GINHeuristic_34608846471453
// MI455X (gfx1250) — matrix operand order repaired and checked by reading, not compiled, not run on hardware
//
#include <hip/hip_runtime.h>

#define N_NODES   50000
#define N_EDGES   800000
#define N_GRAPHS  256
#define IN_DIM    7
#define HID       128
#define EDGE_DIM  4
#define LAYERS    4
#define MTILES    5          // 80 rows per block; 50000 / 80 = 625 exactly

typedef __attribute__((ext_vector_type(16))) _Float16 v16h;
typedef __attribute__((ext_vector_type(8)))  _Float16 v8h;
typedef __attribute__((ext_vector_type(8)))  float    v8f;

#if defined(__has_builtin)
#if __has_builtin(__builtin_amdgcn_sched_group_barrier)
#define SCHED_GROUP(mask, size, id) __builtin_amdgcn_sched_group_barrier((mask), (size), (id))
#endif
#endif
#ifndef SCHED_GROUP
#define SCHED_GROUP(mask, size, id)
#endif

// ---------------------------------------------------------------------------
// Layer-0 edge kernel (d_in = 7): one thread per edge.
// m = relu(x[src] + edge_attr@We + be); aggr[dst] += m  (aggr stride = HID)
// ---------------------------------------------------------------------------
__global__ __launch_bounds__(256) void edge_layer0(
    const long long* __restrict__ ei, const float* __restrict__ ea,
    const float* __restrict__ x, const float* __restrict__ We /*[4][7]*/,
    const float* __restrict__ be, float* __restrict__ aggr, int nE)
{
    int e = blockIdx.x * blockDim.x + threadIdx.x;
    if (e >= nE) return;
    long long s = ei[e];
    long long d = ei[nE + e];
    float a0 = ea[(size_t)e * 4 + 0];
    float a1 = ea[(size_t)e * 4 + 1];
    float a2 = ea[(size_t)e * 4 + 2];
    float a3 = ea[(size_t)e * 4 + 3];
    const float* xr = x + (size_t)s * IN_DIM;
    float* ap = aggr + (size_t)d * HID;
#pragma unroll
    for (int c = 0; c < IN_DIM; ++c) {
        float v = be[c]
                + a0 * We[0 * IN_DIM + c] + a1 * We[1 * IN_DIM + c]
                + a2 * We[2 * IN_DIM + c] + a3 * We[3 * IN_DIM + c]
                + xr[c];
        v = fmaxf(v, 0.0f);
        atomicAdd(ap + c, v);
    }
}

// ---------------------------------------------------------------------------
// Hidden-layer edge kernel (d_in = 128): one wave per edge, float4 per lane.
// ---------------------------------------------------------------------------
__global__ __launch_bounds__(256) void edge_layer_hid(
    const long long* __restrict__ ei, const float* __restrict__ ea,
    const float* __restrict__ h, const float* __restrict__ We /*[4][128]*/,
    const float* __restrict__ be, float* __restrict__ aggr, int nE)
{
    const int lane = threadIdx.x & 31;
    const int e = blockIdx.x * 8 + (threadIdx.x >> 5);
    if (e >= nE) return;
    const long long s = ei[e];
    const long long d = ei[nE + e];
    const float4 a4 = *(const float4*)(ea + (size_t)e * 4);
    const int c = lane * 4;
    const float4 w0 = *(const float4*)(We + 0 * HID + c);
    const float4 w1 = *(const float4*)(We + 1 * HID + c);
    const float4 w2 = *(const float4*)(We + 2 * HID + c);
    const float4 w3 = *(const float4*)(We + 3 * HID + c);
    const float4 bb = *(const float4*)(be + c);
    const float4 hv = *(const float4*)(h + (size_t)s * HID + c);
    float m0 = fmaxf(hv.x + bb.x + a4.x*w0.x + a4.y*w1.x + a4.z*w2.x + a4.w*w3.x, 0.0f);
    float m1 = fmaxf(hv.y + bb.y + a4.x*w0.y + a4.y*w1.y + a4.z*w2.y + a4.w*w3.y, 0.0f);
    float m2 = fmaxf(hv.z + bb.z + a4.x*w0.z + a4.y*w1.z + a4.z*w2.z + a4.w*w3.z, 0.0f);
    float m3 = fmaxf(hv.w + bb.w + a4.x*w0.w + a4.y*w1.w + a4.z*w2.w + a4.w*w3.w, 0.0f);
    float* dp = aggr + (size_t)d * HID + c;
    atomicAdd(dp + 0, m0);
    atomicAdd(dp + 1, m1);
    atomicAdd(dp + 2, m2);
    atomicAdd(dp + 3, m3);
}

// ---------------------------------------------------------------------------
// z = x + aggr (layer 0, pad 7 -> 128 with zeros), cast to f16
// ---------------------------------------------------------------------------
__global__ __launch_bounds__(256) void make_z0(
    const float* __restrict__ x, const float* __restrict__ aggr,
    _Float16* __restrict__ z, int total)
{
    int i = blockIdx.x * blockDim.x + threadIdx.x;
    if (i >= total) return;
    int node = i >> 7, c = i & 127;
    float v = 0.0f;
    if (c < IN_DIM) v = x[(size_t)node * IN_DIM + c] + aggr[i];
    z[i] = (_Float16)v;
}

// z = h + aggr, cast to f16
__global__ __launch_bounds__(256) void make_z(
    const float* __restrict__ h, const float* __restrict__ aggr,
    _Float16* __restrict__ z, int total)
{
    int i = blockIdx.x * blockDim.x + threadIdx.x;
    if (i >= total) return;
    z[i] = (_Float16)(h[i] + aggr[i]);
}

// ---------------------------------------------------------------------------
// Transpose + cast weight W[kact][128] (f32) -> WT[128][128] f16 (zero-padded K)
// WT[n][k] = W[k][n]
// ---------------------------------------------------------------------------
__global__ __launch_bounds__(256) void prep_wt(
    const float* __restrict__ W, int kact, _Float16* __restrict__ WT)
{
    int i = blockIdx.x * blockDim.x + threadIdx.x;
    if (i >= HID * HID) return;
    int n = i >> 7, k = i & 127;
    float v = (k < kact) ? W[(size_t)k * HID + n] : 0.0f;
    WT[(size_t)n * HID + k] = (_Float16)v;
}

// ---------------------------------------------------------------------------
// out = relu(A[50000,128] @ W[128,128] + bias) via v_wmma_f32_16x16x32_f16.
// Block: 80 rows (5 M-tiles) x 8 waves (one 16-col tile per wave).
// Per K-step: 12 b128 fragment loads (1 B frag + 5 A frags), then 5 WMMAs
// sharing the B fragment. sched_group_barrier pins the shape
// (12 VMEM-reads -> 5 WMMA) so pre-RA scheduling keeps all loads ahead of the
// WMMA burst, forcing distinct registers and a single s_wait_loadcnt.
// Fragment layouts (ISA 7.12.2, 16-bit):
//   A: M = lane%16; elems 0..7 -> K=k0+half*8+i, elems 8..15 -> K=k0+16+half*8+i
//   B: N = lane%16; elems 0..7 -> K=k0+half*8+i, elems 8..15 -> K=k0+16+half*8+i
//   C/D: VGPR r -> row M = r + 8*half; col N = lane%16
// 50000 % 80 == 0 -> no bounds checks; OUT_F16 resolved at compile time.
// ---------------------------------------------------------------------------
template <bool OUT_F16>
__global__ __launch_bounds__(256) void gemm128_relu(
    const _Float16* __restrict__ A, const _Float16* __restrict__ WT,
    const float* __restrict__ bias, void* __restrict__ out)
{
    const int lane = threadIdx.x & 31;
    const int wave = threadIdx.x >> 5;   // 8 column tiles
    const int half = lane >> 4;
    const int l16  = lane & 15;
    const int m0 = blockIdx.x * (16 * MTILES);
    const int n0 = wave * 16;

    const _Float16* arow = A  + (size_t)(m0 + l16) * HID;
    const _Float16* wcol = WT + (size_t)(n0 + l16) * HID;

    v8f acc[MTILES];
#pragma unroll
    for (int t = 0; t < MTILES; ++t) acc[t] = (v8f){};

#pragma unroll
    for (int k0 = 0; k0 < HID; k0 += 32) {
        // --- load phase: 12 independent b128 loads -------------------------
        v8h b0 = *(const v8h*)(wcol + k0 + half * 8);
        v8h b1 = *(const v8h*)(wcol + k0 + 16 + half * 8);
        v8h alo[MTILES], ahi[MTILES];
#pragma unroll
        for (int t = 0; t < MTILES; ++t) {
            const _Float16* ar = arow + (size_t)t * 16 * HID;
            alo[t] = *(const v8h*)(ar + k0 + half * 8);
            ahi[t] = *(const v8h*)(ar + k0 + 16 + half * 8);
        }
        // --- compute phase: 5 WMMAs sharing one B fragment -----------------
        v16h b;
#pragma unroll
        for (int i = 0; i < 8; ++i) { b[i] = b0[i]; b[8 + i] = b1[i]; }
#pragma unroll
        for (int t = 0; t < MTILES; ++t) {
            v16h a;
#pragma unroll
            for (int i = 0; i < 8; ++i) { a[i] = alo[t][i]; a[8 + i] = ahi[t][i]; }
            acc[t] = __builtin_amdgcn_wmma_f32_16x16x32_f16(
                         /*neg_a=*/false, a, /*neg_b=*/false, b,
                         /*c_mod=*/(short)0, acc[t],
                         /*reuse_a=*/false, /*reuse_b=*/false);
        }
        // Desired per-K-step pipeline: all VMEM reads first, then the WMMAs.
        SCHED_GROUP(0x020, 12, 0);   // 12 VMEM read instructions
        SCHED_GROUP(0x008,  5, 0);   // 5 WMMA (MFMA-class) instructions
    }

    const int col = n0 + l16;
    const float bv = bias[col];
#pragma unroll
    for (int t = 0; t < MTILES; ++t) {
#pragma unroll
        for (int r = 0; r < 8; ++r) {
            const int row = m0 + t * 16 + half * 8 + r;
            const float v = fmaxf(acc[t][r] + bv, 0.0f);
            if (OUT_F16)
                ((_Float16*)out)[(size_t)row * HID + col] = (_Float16)v;
            else
                ((float*)out)[(size_t)row * HID + col] = v;
        }
    }
}

// ---------------------------------------------------------------------------
// Per-graph sum pool + counts (atomics), then head MLP per graph.
// ---------------------------------------------------------------------------
__global__ __launch_bounds__(256) void pool_kernel(
    const float* __restrict__ h, const long long* __restrict__ batch,
    float* __restrict__ hg, float* __restrict__ cnt, int nN)
{
    int i = blockIdx.x * blockDim.x + threadIdx.x;
    if (i >= nN * HID) return;
    int node = i >> 7, c = i & 127;
    int g = (int)batch[node];
    atomicAdd(&hg[(size_t)g * HID + c], h[i]);
    if (c == 0) atomicAdd(&cnt[g], 1.0f);
}

__global__ __launch_bounds__(128) void head_kernel(
    const float* __restrict__ hg, const float* __restrict__ cnt,
    const float* __restrict__ W1, const float* __restrict__ b1,
    const float* __restrict__ W2, const float* __restrict__ b2,
    float* __restrict__ out)
{
    __shared__ float hm[HID];
    __shared__ float red[HID];
    const int g = blockIdx.x, t = threadIdx.x;
    const float cv = fmaxf(cnt[g], 1.0f);
    hm[t] = hg[(size_t)g * HID + t] / cv;
    __syncthreads();
    float acc = b1[t];
    for (int k = 0; k < HID; ++k) acc += hm[k] * W1[(size_t)k * HID + t];
    acc = fmaxf(acc, 0.0f);
    red[t] = acc * W2[t];
    __syncthreads();
    for (int s = 64; s > 0; s >>= 1) {
        if (t < s) red[t] += red[t + s];
        __syncthreads();
    }
    if (t == 0) out[g] = red[0] + b2[0];
}

// ---------------------------------------------------------------------------
// Orchestration
// ---------------------------------------------------------------------------
extern "C" void kernel_launch(void* const* d_in, const int* in_sizes, int n_in,
                              void* d_out, int out_size, void* d_ws, size_t ws_size,
                              hipStream_t stream)
{
    (void)in_sizes; (void)n_in; (void)out_size; (void)ws_size;

    const float*     x     = (const float*)d_in[0];
    const long long* ei    = (const long long*)d_in[1];
    const long long* batch = (const long long*)d_in[2];
    const float*     ea    = (const float*)d_in[3];
    // conv_params flattened: per layer {We, be, W1, b1, W2, b2, W3, b3} at 4 + L*8
    const float* Wh1 = (const float*)d_in[4 + LAYERS * 8 + 0];
    const float* bh1 = (const float*)d_in[4 + LAYERS * 8 + 1];
    const float* Wh2 = (const float*)d_in[4 + LAYERS * 8 + 2];
    const float* bh2 = (const float*)d_in[4 + LAYERS * 8 + 3];

    // workspace carve (256B aligned)
    char* p = (char*)d_ws;
    auto carve = [&](size_t bytes) -> void* {
        void* r = (void*)p;
        p += (bytes + 255) & ~(size_t)255;
        return r;
    };
    float*    h    = (float*)carve((size_t)N_NODES * HID * sizeof(float));
    float*    aggr = (float*)carve((size_t)N_NODES * HID * sizeof(float));
    _Float16* zf   = (_Float16*)carve((size_t)N_NODES * HID * sizeof(_Float16));
    _Float16* t1   = (_Float16*)carve((size_t)N_NODES * HID * sizeof(_Float16));
    _Float16* wt   = (_Float16*)carve((size_t)HID * HID * sizeof(_Float16));
    float*    hg   = (float*)carve((size_t)N_GRAPHS * HID * sizeof(float));
    float*    cnt  = (float*)carve((size_t)N_GRAPHS * sizeof(float));

    const int zBlocks    = (N_NODES * HID + 255) / 256;
    const int wtBlocks   = (HID * HID + 255) / 256;
    const int gemmBlocks = N_NODES / (16 * MTILES);   // 625, exact

    for (int L = 0; L < LAYERS; ++L) {
        const float* We = (const float*)d_in[4 + L * 8 + 0];
        const float* be = (const float*)d_in[4 + L * 8 + 1];
        const float* W1 = (const float*)d_in[4 + L * 8 + 2];
        const float* b1 = (const float*)d_in[4 + L * 8 + 3];
        const float* W2 = (const float*)d_in[4 + L * 8 + 4];
        const float* b2 = (const float*)d_in[4 + L * 8 + 5];
        const float* W3 = (const float*)d_in[4 + L * 8 + 6];
        const float* b3 = (const float*)d_in[4 + L * 8 + 7];
        const int kact = (L == 0) ? IN_DIM : HID;

        hipMemsetAsync(aggr, 0, (size_t)N_NODES * HID * sizeof(float), stream);
        if (L == 0)
            edge_layer0<<<(N_EDGES + 255) / 256, 256, 0, stream>>>(ei, ea, x, We, be, aggr, N_EDGES);
        else
            edge_layer_hid<<<(N_EDGES + 7) / 8, 256, 0, stream>>>(ei, ea, h, We, be, aggr, N_EDGES);

        if (L == 0)
            make_z0<<<zBlocks, 256, 0, stream>>>(x, aggr, zf, N_NODES * HID);
        else
            make_z<<<zBlocks, 256, 0, stream>>>(h, aggr, zf, N_NODES * HID);

        prep_wt<<<wtBlocks, 256, 0, stream>>>(W1, kact, wt);
        gemm128_relu<true><<<gemmBlocks, 256, 0, stream>>>(zf, wt, b1, (void*)t1);
        prep_wt<<<wtBlocks, 256, 0, stream>>>(W2, HID, wt);
        gemm128_relu<true><<<gemmBlocks, 256, 0, stream>>>(t1, wt, b2, (void*)zf);
        prep_wt<<<wtBlocks, 256, 0, stream>>>(W3, HID, wt);
        gemm128_relu<false><<<gemmBlocks, 256, 0, stream>>>(zf, wt, b3, (void*)h);
    }

    hipMemsetAsync(hg, 0, (size_t)N_GRAPHS * HID * sizeof(float), stream);
    hipMemsetAsync(cnt, 0, (size_t)N_GRAPHS * sizeof(float), stream);
    pool_kernel<<<zBlocks, 256, 0, stream>>>(h, batch, hg, cnt, N_NODES);
    head_kernel<<<N_GRAPHS, 128, 0, stream>>>(hg, cnt, Wh1, bh1, Wh2, bh2, (float*)d_out);
}
